// LagrangeInterp_RNN_66013647340233
// MI455X (gfx1250) — matrix operand order repaired and checked by reading, not compiled, not run on hardware
//
#include <hip/hip_runtime.h>
#include <hip/hip_bf16.h>
#include <stdint.h>

#define HIDDEN 1024
#define INPUT  512
#define BATCH  64
#define TSTEPS 512
#define GATES  (4 * HIDDEN)       // 4096
#define KTOT   (INPUT + HIDDEN)   // 1536
#define APAD   8                  // bf16 row padding -> stride 772 dwords (≡4 mod 64 banks)
#define AROW   (KTOT + APAD)      // 1544 bf16 per LDS row

typedef __bf16 bf16_t;
typedef __attribute__((ext_vector_type(16))) __bf16 v16bf;
typedef __attribute__((ext_vector_type(8)))  __bf16 v8bf;
typedef __attribute__((ext_vector_type(4)))  __bf16 v4bf;
typedef __attribute__((ext_vector_type(8)))  float  v8f;

// Lagrange kernel, delta = OS_FACTOR-1 = 0.5, order 3 (compile-time constants)
#define LK0 (0.3125f)
#define LK1 (0.9375f)
#define LK2 (-0.3125f)
#define LK3 (0.0625f)

// ---- workspace layout (bytes) --------------------------------------------
#define OFF_XB    ((size_t)0)           // bf16 x:    B*T*INPUT*2   = 32 MiB
#define OFF_WCAT  ((size_t)33554432)    // bf16 Wcat: GATES*KTOT*2  = 12 MiB
#define OFF_RING  ((size_t)46137344)    // f32 ring:  4*B*2H*4      =  2 MiB
#define OFF_HREAD ((size_t)48234496)    // bf16 hread: B*H*2        = 128 KiB
#define OFF_CPREV ((size_t)48365568)    // f32 cprev:  B*H*4        = 256 KiB
#define OFF_GATES ((size_t)48627712)    // f32 gates:  B*4H*4       =   1 MiB

// ---- one-time conversion kernels -----------------------------------------
__global__ __launch_bounds__(256) void cvt_x_kernel(const float4* __restrict__ x,
                                                    bf16_t* __restrict__ xb) {
    int i = blockIdx.x * 256 + threadIdx.x;   // B*T*INPUT/4 threads
    float4 v = x[i];
    v4bf o;
    o.x = (bf16_t)v.x; o.y = (bf16_t)v.y; o.z = (bf16_t)v.z; o.w = (bf16_t)v.w;
    *(v4bf*)(xb + (size_t)i * 4) = o;
}

// Pack [W_ih | W_hh] row-major by gate: Wcat[n][k], k<512 -> W_ih, else W_hh
__global__ __launch_bounds__(256) void pack_w_kernel(const float* __restrict__ Wih,
                                                     const float* __restrict__ Whh,
                                                     bf16_t* __restrict__ Wcat) {
    int idx = blockIdx.x * 256 + threadIdx.x;   // GATES*KTOT threads
    int n = idx / KTOT;
    int k = idx - n * KTOT;
    float v = (k < INPUT) ? Wih[(size_t)n * INPUT + k]
                          : Whh[(size_t)n * HIDDEN + (k - INPUT)];
    Wcat[idx] = (bf16_t)v;
}

// ---- per-step GEMM: gates[b][n] = bias[n] + [x_t | h_read] @ Wcat[n][:] --
// Block = 8 waves, all sharing one mtile; A panel staged once into LDS via
// the CDNA5 async-to-LDS path, B streamed from L2-resident packed weights.
__global__ __launch_bounds__(256) void gemm_step_kernel(
    const bf16_t* __restrict__ xb,      // [B][T][INPUT] bf16
    const bf16_t* __restrict__ hread,   // [B][HIDDEN]  bf16
    const bf16_t* __restrict__ Wcat,    // [GATES][KTOT] bf16
    const float*  __restrict__ b_ih,
    const float*  __restrict__ b_hh,
    float*        __restrict__ gates,   // [B][GATES] f32
    int t) {
    __shared__ bf16_t Alds[16 * AROW];  // 16 rows x 1544 bf16 = 48.25 KiB

    const int tidb = threadIdx.x;
    const int lane = tidb & 31;
    const int wave = tidb >> 5;
    const int gid  = blockIdx.x * 8 + wave;   // 0..1023 wave tiles
    const int mtile = gid >> 8;               // 0..3 (identical for all 8 waves)
    const int ntile = gid & 255;              // 0..255 (gate tiles of 16)
    const int r    = lane & 15;
    const int half = lane >> 4;

    // ---- cooperative async stage of A panel: 16 x 1536 bf16, 16B chunks ----
    // 192 chunks per row, 3072 chunks total, 12 per thread.
    const uint32_t lds_base = (uint32_t)(uintptr_t)Alds;  // addr[31:0] == LDS offset
    #pragma unroll
    for (int i = 0; i < 12; ++i) {
        const int c   = tidb + i * 256;
        const int row = c / 192;
        const int cw  = c - row * 192;          // 16B chunk within row
        const int rg  = mtile * 16 + row;       // global batch row
        const bf16_t* gsrc = (cw < 64)
            ? (xb + ((size_t)rg * TSTEPS + t) * INPUT + (size_t)cw * 8)
            : (hread + (size_t)rg * HIDDEN + ((size_t)cw * 8 - INPUT));
        const uint32_t ldsa = lds_base + (uint32_t)(row * AROW + cw * 8) * 2u;
        asm volatile("global_load_async_to_lds_b128 %0, %1, off"
                     :: "v"(ldsa), "v"((uint64_t)(uintptr_t)gsrc)
                     : "memory");
    }
#if __has_builtin(__builtin_amdgcn_s_wait_asynccnt)
    __builtin_amdgcn_s_wait_asynccnt(0);
#else
    asm volatile("s_wait_asynccnt 0x0" ::: "memory");
#endif
    __syncthreads();

    // ---- WMMA main loop: A from LDS (ds_load_b128), B from global ----------
    const int colB = ntile * 16 + r;                       // gate col for B / D
    const bf16_t* __restrict__ Arow = Alds + (size_t)r * AROW;
    const bf16_t* __restrict__ Brow = Wcat + (size_t)colB * KTOT;

    v8f acc = {};
    #pragma unroll 4
    for (int kc = 0; kc < KTOT; kc += 32) {
        v8bf alo = *(const v8bf*)(Arow + kc + half * 16);      // K = kc+half*16 ..
        v8bf ahi = *(const v8bf*)(Arow + kc + half * 16 + 8);  // K = kc+half*16+8 ..
        v16bf a = __builtin_shufflevector(alo, ahi,
                                          0,1,2,3,4,5,6,7,8,9,10,11,12,13,14,15);
        v16bf b = *(const v16bf*)(Brow + kc + half * 16);      // K = kc+half*16 ..
        acc = __builtin_amdgcn_wmma_f32_16x16x32_bf16(false, a, false, b,
                                                      (short)0, acc, false, false);
    }

    const float bias = b_ih[colB] + b_hh[colB];
    #pragma unroll
    for (int v = 0; v < 8; ++v) {
        const int row = half * 8 + v;                          // D layout: M = half*8+v
        gates[(size_t)(mtile * 16 + row) * GATES + colB] = acc[v] + bias;
    }
}

// ---- per-step pointwise: LSTM math + ring push + next Lagrange read ------
__global__ __launch_bounds__(256) void pointwise_step_kernel(
    const float* __restrict__ gates,   // [B][GATES]
    float*       __restrict__ ring,    // [4][B][2H]
    bf16_t*      __restrict__ hread,   // [B][H] (next-step bf16 A operand)
    float*       __restrict__ cprev,   // [B][H]
    float*       __restrict__ out_h,   // [B][T][H]
    float*       __restrict__ out_last,// [B][2H]
    int t) {
    const int tid = blockIdx.x * 256 + threadIdx.x;   // B*H threads
    const int b = tid >> 10;
    const int j = tid & (HIDDEN - 1);

    const float* g = gates + (size_t)b * GATES;
    const float ig = g[j];
    const float fg = g[HIDDEN + j];
    const float gg = g[2 * HIDDEN + j];
    const float og = g[3 * HIDDEN + j];

    const float si = 1.0f / (1.0f + __expf(-ig));
    const float sf = 1.0f / (1.0f + __expf(-fg));
    const float so = 1.0f / (1.0f + __expf(-og));
    const float c  = sf * cprev[tid] + si * tanhf(gg);
    const float h  = so * tanhf(c);

    out_h[((size_t)b * TSTEPS + t) * HIDDEN + j] = h;
    if (t == TSTEPS - 1) {
        out_last[(size_t)b * 2 * HIDDEN + j]          = h;
        out_last[(size_t)b * 2 * HIDDEN + HIDDEN + j] = c;
    }

    // circular ring push (slot = t&3); zero-init covers t-k < 0 entries
    const int slot = t & 3;
    float* Rw = ring + ((size_t)slot * BATCH + b) * 2 * HIDDEN;
    Rw[j]          = h;
    Rw[HIDDEN + j] = c;

    const float* R1 = ring + ((size_t)((t - 1) & 3) * BATCH + b) * 2 * HIDDEN;
    const float* R2 = ring + ((size_t)((t - 2) & 3) * BATCH + b) * 2 * HIDDEN;
    const float* R3 = ring + ((size_t)((t - 3) & 3) * BATCH + b) * 2 * HIDDEN;

    const float hr = LK0 * h + LK1 * R1[j] + LK2 * R2[j] + LK3 * R3[j];
    const float cr = LK0 * c + LK1 * R1[HIDDEN + j] + LK2 * R2[HIDDEN + j]
                   + LK3 * R3[HIDDEN + j];

    hread[tid] = (bf16_t)hr;
    cprev[tid] = cr;
}

extern "C" void kernel_launch(void* const* d_in, const int* in_sizes, int n_in,
                              void* d_out, int out_size, void* d_ws, size_t ws_size,
                              hipStream_t stream) {
    const float* x    = (const float*)d_in[0];
    const float* W_ih = (const float*)d_in[1];
    const float* W_hh = (const float*)d_in[2];
    const float* b_ih = (const float*)d_in[3];
    const float* b_hh = (const float*)d_in[4];
    float* out = (float*)d_out;
    char*  ws  = (char*)d_ws;

    bf16_t* xb    = (bf16_t*)(ws + OFF_XB);
    bf16_t* Wcat  = (bf16_t*)(ws + OFF_WCAT);
    float*  ring  = (float*) (ws + OFF_RING);
    bf16_t* hread = (bf16_t*)(ws + OFF_HREAD);
    float*  cprev = (float*) (ws + OFF_CPREV);
    float*  gates = (float*) (ws + OFF_GATES);

    // deterministic zero-init of scan state every call (ws is poisoned once)
    hipMemsetAsync(ring,  0, (size_t)4 * BATCH * 2 * HIDDEN * sizeof(float), stream);
    hipMemsetAsync(hread, 0, (size_t)BATCH * HIDDEN * sizeof(bf16_t), stream);
    hipMemsetAsync(cprev, 0, (size_t)BATCH * HIDDEN * sizeof(float), stream);

    // one-time (per call) bf16 conversion / weight packing
    cvt_x_kernel<<<(BATCH * TSTEPS * INPUT) / (256 * 4), 256, 0, stream>>>(
        (const float4*)x, xb);
    pack_w_kernel<<<(GATES * KTOT) / 256, 256, 0, stream>>>(W_ih, W_hh, Wcat);

    float* out_h    = out;
    float* out_last = out + (size_t)BATCH * TSTEPS * HIDDEN;

    for (int t = 0; t < TSTEPS; ++t) {
        gemm_step_kernel<<<128, 256, 0, stream>>>(xb, hread, Wcat, b_ih, b_hh,
                                                  gates, t);
        pointwise_step_kernel<<<BATCH * HIDDEN / 256, 256, 0, stream>>>(
            gates, ring, hread, cprev, out_h, out_last, t);
    }
}
